// PolarGate_37744172597711
// MI455X (gfx1250) — matrix operand order repaired and checked by reading, not compiled, not run on hardware
//
#include <hip/hip_runtime.h>
#include <hip/hip_bf16.h>
#include <math.h>

// ---------------------------------------------------------------------------
// PolarGate-style signed GNN forward for MI455X (gfx1250, wave32, WMMA).
// Edge aggregation: L2-resident float atomics (192MB L2 >> 25MB node arrays).
// Node GEMMs: v_wmma_f32_16x16x32_bf16 (f32 accumulate), fragments per
// CDNA5 ISA §7.12.2 layouts.
// ---------------------------------------------------------------------------

typedef __attribute__((ext_vector_type(16))) __bf16 v16bf;
typedef __attribute__((ext_vector_type(8)))  float  v8f;

__device__ __forceinline__ v8f wmma_bf16(v16bf a, v16bf b, v8f c) {
  // (neg_a, A, neg_b, B, c_mod, C, reuse_a, reuse_b)
  return __builtin_amdgcn_wmma_f32_16x16x32_bf16(false, a, false, b, (short)0, c,
                                                 false, false);
}

__device__ __forceinline__ v8f zero8() {
  v8f z;
#pragma unroll
  for (int i = 0; i < 8; ++i) z[i] = 0.0f;
  return z;
}

// A-fragment (16x32, bf16) for one 32-wide K chunk. Per §7.12.2:
// lane<16 holds row M=lane, K = {0..7, 16..23}; lane>=16 holds same row,
// K = {8..15, 24..31}. p = per-lane row pointer (32 consecutive f32), scaled.
__device__ __forceinline__ v16bf a_frag32(const float* p, float s, int lane) {
  const int kb = (lane & 16) ? 8 : 0;
  v16bf a;
#pragma unroll
  for (int i = 0; i < 8; ++i) a[i] = (__bf16)(p[kb + i] * s);
#pragma unroll
  for (int i = 0; i < 8; ++i) a[8 + i] = (__bf16)(p[kb + 16 + i] * s);
  return a;
}

// B-fragment (32x16, bf16) from row-major KxN weight tile (leading dim ldn):
// lane n<16 holds K = k0+{0..7, 16..23} for column n0+n;
// lanes 16..31 hold K = k0+{8..15, 24..31} for the same columns.
__device__ __forceinline__ v16bf b_frag(const __bf16* W, int ldn, int k0, int n0,
                                        int lane) {
  const int col = n0 + (lane & 15);
  const int kb  = k0 + ((lane & 16) ? 8 : 0);
  v16bf b;
#pragma unroll
  for (int i = 0; i < 8; ++i) b[i] = W[(kb + i) * ldn + col];
#pragma unroll
  for (int i = 0; i < 8; ++i) b[8 + i] = W[(kb + 16 + i) * ldn + col];
  return b;
}

// ---------------------------------------------------------------------------
// Zero fill (grid-stride).
__global__ void zero_kernel(float* __restrict__ p, size_t n) {
  size_t i = (size_t)blockIdx.x * blockDim.x + threadIdx.x;
  const size_t stride = (size_t)gridDim.x * blockDim.x;
  for (; i < n; i += stride) p[i] = 0.0f;
}

// ---------------------------------------------------------------------------
// Edge gather->scatter-add. 16 threads per edge, each moves 4 floats of the
// 64-wide feature row (float4 gather, 4x global_atomic_add_f32 into L2).
// Positive-sign edges accumulate into accP, negative into accN.
__global__ void edge_agg(const int* __restrict__ ei, const float* __restrict__ x,
                         float* accP, float* accN, float* cntP, float* cntN,
                         int E, int addCount) {
  const long long t = (long long)blockIdx.x * blockDim.x + threadIdx.x;
  const long long e = t >> 4;
  const int part    = (int)(t & 15);
  if (e >= E) return;
  const int s  = ei[3 * e + 0];
  const int d  = ei[3 * e + 1];
  const int sg = ei[3 * e + 2];
  float* acc = (sg > 0) ? accP : accN;
  const float4 v = *(const float4*)(x + (size_t)s * 64 + part * 4);
  float* q = acc + (size_t)d * 64 + part * 4;
  atomicAdd(q + 0, v.x);
  atomicAdd(q + 1, v.y);
  atomicAdd(q + 2, v.z);
  atomicAdd(q + 3, v.w);
  if (addCount && part == 0) atomicAdd(((sg > 0) ? cntP : cntN) + d, 1.0f);
}

// ---------------------------------------------------------------------------
// Layer 1: z1 = tanh([ [ap,x]@Wp1+bp1 , [an,x]@Wn1+bn1 ]), 16-row tile/wave.
__global__ void __launch_bounds__(128)
node_layer1(const float* __restrict__ x, const float* __restrict__ accP,
            const float* __restrict__ accN, const float* __restrict__ cntP,
            const float* __restrict__ cntN, const float* __restrict__ Wp1,
            const float* __restrict__ bp1, const float* __restrict__ Wn1,
            const float* __restrict__ bn1, float* __restrict__ z1, int N,
            int nTiles) {
  __shared__ __bf16 sWp[128 * 32];
  __shared__ __bf16 sWn[128 * 32];
  for (int i = threadIdx.x; i < 128 * 32; i += blockDim.x) {
    sWp[i] = (__bf16)Wp1[i];
    sWn[i] = (__bf16)Wn1[i];
  }
  __syncthreads();

  const int lane = threadIdx.x & 31;
  const int gw   = blockIdx.x * (blockDim.x >> 5) + (threadIdx.x >> 5);
  const int nw   = gridDim.x * (blockDim.x >> 5);

  for (int tile = gw; tile < nTiles; tile += nw) {
    int row = tile * 16 + (lane & 15);
    if (row >= N) row = N - 1;
    const float* xr  = x + (size_t)row * 64;
    const float* apr = accP + (size_t)row * 64;
    const float* anr = accN + (size_t)row * 64;
    const float icp = 1.0f / fmaxf(cntP[row], 1.0f);
    const float icn = 1.0f / fmaxf(cntN[row], 1.0f);

    // A rows are [agg(64) | x(64)]: each 32-wide K chunk maps wholly to one.
    v16bf ap[4], an[4];
#pragma unroll
    for (int c = 0; c < 2; ++c) {
      ap[c] = a_frag32(apr + c * 32, icp, lane);
      an[c] = a_frag32(anr + c * 32, icn, lane);
    }
#pragma unroll
    for (int c = 0; c < 2; ++c) {
      const v16bf xf = a_frag32(xr + c * 32, 1.0f, lane);
      ap[2 + c] = xf;
      an[2 + c] = xf;
    }

#pragma unroll
    for (int nt = 0; nt < 2; ++nt) {
      v8f cp = zero8(), cn = zero8();
#pragma unroll
      for (int c = 0; c < 4; ++c) {
        cp = wmma_bf16(ap[c], b_frag(sWp, 32, c * 32, nt * 16, lane), cp);
        cn = wmma_bf16(an[c], b_frag(sWn, 32, c * 32, nt * 16, lane), cn);
      }
      const int col = nt * 16 + (lane & 15);
      const int rb  = (lane & 16) ? 8 : 0;
      const float bP = bp1[col], bN = bn1[col];
#pragma unroll
      for (int r = 0; r < 8; ++r) {
        const int nrow = tile * 16 + rb + r;
        if (nrow < N) {
          z1[(size_t)nrow * 64 + col]      = tanhf(cp[r] + bP);  // xp half
          z1[(size_t)nrow * 64 + 32 + col] = tanhf(cn[r] + bN);  // xn half
        }
      }
    }
  }
}

// ---------------------------------------------------------------------------
// Layer 2 fused: hp/hn (K=96) -> tanh -> @Ww -> tanh (z out) -> BN/ReLU MLP
// (Wm1, Wm2) -> sigmoid dot (Wm3) = prob. LDS tiles give the A-layout
// transpose between chained GEMMs; all LDS traffic is intra-wave (in-order).
#define L2_WAVES 2
__global__ void __launch_bounds__(64)
node_layer2(const float* __restrict__ z1, const float* __restrict__ accP,
            const float* __restrict__ accN, const float* __restrict__ cntP,
            const float* __restrict__ cntN, const float* __restrict__ Wp2,
            const float* __restrict__ bp2, const float* __restrict__ Wn2,
            const float* __restrict__ bn2, const float* __restrict__ Ww,
            const float* __restrict__ bw, const float* __restrict__ Wm1,
            const float* __restrict__ bm1, const float* __restrict__ g1,
            const float* __restrict__ be1, const float* __restrict__ rm1,
            const float* __restrict__ rv1, const float* __restrict__ Wm2,
            const float* __restrict__ bm2, const float* __restrict__ g2,
            const float* __restrict__ be2, const float* __restrict__ rm2,
            const float* __restrict__ rv2, const float* __restrict__ Wm3,
            const float* __restrict__ bm3, float* __restrict__ outZ,
            float* __restrict__ outProb, int N, int nTiles) {
  __shared__ __bf16 sWp2[96 * 32];
  __shared__ __bf16 sWn2[96 * 32];
  __shared__ __bf16 sWw[64 * 64];
  __shared__ __bf16 sWm1[64 * 64];
  __shared__ __bf16 sWm2[64 * 64];
  __shared__ float  tAs[L2_WAVES][16 * 64];
  __shared__ float  tBs[L2_WAVES][16 * 64];

  for (int i = threadIdx.x; i < 96 * 32; i += blockDim.x) {
    sWp2[i] = (__bf16)Wp2[i];
    sWn2[i] = (__bf16)Wn2[i];
  }
  for (int i = threadIdx.x; i < 64 * 64; i += blockDim.x) {
    sWw[i]  = (__bf16)Ww[i];
    sWm1[i] = (__bf16)Wm1[i];
    sWm2[i] = (__bf16)Wm2[i];
  }
  __syncthreads();

  const int lane = threadIdx.x & 31;
  const int wv   = threadIdx.x >> 5;
  float* ta = tAs[wv];
  float* tb = tBs[wv];
  const int gw = blockIdx.x * L2_WAVES + wv;
  const int nw = gridDim.x * L2_WAVES;
  const int col16 = lane & 15;
  const int rb    = (lane & 16) ? 8 : 0;

  for (int tile = gw; tile < nTiles; tile += nw) {
    int row = tile * 16 + col16;
    if (row >= N) row = N - 1;
    const float icp = 1.0f / fmaxf(cntP[row], 1.0f);
    const float icn = 1.0f / fmaxf(cntN[row], 1.0f);
    const float* pP = accP + (size_t)row * 64;  // [aggP(xp) | aggP(xn)]
    const float* pN = accN + (size_t)row * 64;  // [aggN(xp) | aggN(xn)]
    const float* pz = z1 + (size_t)row * 64;    // [xp | xn]

    v16bf aHp[3], aHn[3];
    aHp[0] = a_frag32(pP + 0, icp, lane);   // mean_agg(xp, pos)
    aHp[1] = a_frag32(pN + 32, icn, lane);  // mean_agg(xn, neg)
    aHp[2] = a_frag32(pz + 0, 1.0f, lane);  // xp
    aHn[0] = a_frag32(pP + 32, icp, lane);  // mean_agg(xn, pos)
    aHn[1] = a_frag32(pN + 0, icn, lane);   // mean_agg(xp, neg)
    aHn[2] = a_frag32(pz + 32, 1.0f, lane); // xn

    // z2 = tanh([hp|hn]) -> ta
#pragma unroll
    for (int nt = 0; nt < 2; ++nt) {
      v8f cp = zero8(), cn = zero8();
#pragma unroll
      for (int c = 0; c < 3; ++c) {
        cp = wmma_bf16(aHp[c], b_frag(sWp2, 32, c * 32, nt * 16, lane), cp);
        cn = wmma_bf16(aHn[c], b_frag(sWn2, 32, c * 32, nt * 16, lane), cn);
      }
      const int col = nt * 16 + col16;
      const float bP = bp2[col], bN = bn2[col];
#pragma unroll
      for (int r = 0; r < 8; ++r) {
        ta[(rb + r) * 64 + col]      = tanhf(cp[r] + bP);
        ta[(rb + r) * 64 + 32 + col] = tanhf(cn[r] + bN);
      }
    }

    // z = tanh(z2 @ Ww + bw) -> tb and global output
#pragma unroll
    for (int nt = 0; nt < 4; ++nt) {
      v8f c = zero8();
#pragma unroll
      for (int kc = 0; kc < 2; ++kc)
        c = wmma_bf16(a_frag32(ta + col16 * 64 + kc * 32, 1.0f, lane),
                      b_frag(sWw, 64, kc * 32, nt * 16, lane), c);
      const int col = nt * 16 + col16;
      const float bb = bw[col];
#pragma unroll
      for (int r = 0; r < 8; ++r) {
        const float v = tanhf(c[r] + bb);
        tb[(rb + r) * 64 + col] = v;
        const int nrow = tile * 16 + rb + r;
        if (nrow < N) outZ[(size_t)nrow * 64 + col] = v;
      }
    }

    // h1 = relu(bn1(z @ Wm1 + bm1)) -> ta
#pragma unroll
    for (int nt = 0; nt < 4; ++nt) {
      v8f c = zero8();
#pragma unroll
      for (int kc = 0; kc < 2; ++kc)
        c = wmma_bf16(a_frag32(tb + col16 * 64 + kc * 32, 1.0f, lane),
                      b_frag(sWm1, 64, kc * 32, nt * 16, lane), c);
      const int col = nt * 16 + col16;
      const float sc = rsqrtf(rv1[col] + 1e-5f) * g1[col];
      const float sh = be1[col] - rm1[col] * sc;
      const float bb = bm1[col];
#pragma unroll
      for (int r = 0; r < 8; ++r)
        ta[(rb + r) * 64 + col] = fmaxf((c[r] + bb) * sc + sh, 0.0f);
    }

    // h2 = relu(bn2(h1 @ Wm2 + bm2)) -> tb
#pragma unroll
    for (int nt = 0; nt < 4; ++nt) {
      v8f c = zero8();
#pragma unroll
      for (int kc = 0; kc < 2; ++kc)
        c = wmma_bf16(a_frag32(ta + col16 * 64 + kc * 32, 1.0f, lane),
                      b_frag(sWm2, 64, kc * 32, nt * 16, lane), c);
      const int col = nt * 16 + col16;
      const float sc = rsqrtf(rv2[col] + 1e-5f) * g2[col];
      const float sh = be2[col] - rm2[col] * sc;
      const float bb = bm2[col];
#pragma unroll
      for (int r = 0; r < 8; ++r)
        tb[(rb + r) * 64 + col] = fmaxf((c[r] + bb) * sc + sh, 0.0f);
    }

    // prob = sigmoid(h2 @ Wm3 + bm3): one row per lane 0..15
    if (lane < 16) {
      const int nrow = tile * 16 + lane;
      if (nrow < N) {
        float s = bm3[0];
#pragma unroll
        for (int c2 = 0; c2 < 64; ++c2) s += tb[lane * 64 + c2] * Wm3[c2];
        outProb[nrow] = 1.0f / (1.0f + expf(-s));
      }
    }
  }
}

// ---------------------------------------------------------------------------
static inline int grid_for(size_t n, int cap) {
  size_t b = (n + 255) / 256;
  return (int)(b < (size_t)cap ? b : (size_t)cap);
}

extern "C" void kernel_launch(void* const* d_in, const int* in_sizes, int n_in,
                              void* d_out, int out_size, void* d_ws,
                              size_t ws_size, hipStream_t stream) {
  const float* x   = (const float*)d_in[0];
  const int*   ei  = (const int*)d_in[1];
  const float* Wp1 = (const float*)d_in[2];
  const float* bp1 = (const float*)d_in[3];
  const float* Wn1 = (const float*)d_in[4];
  const float* bn1 = (const float*)d_in[5];
  const float* Wp2 = (const float*)d_in[6];
  const float* bp2 = (const float*)d_in[7];
  const float* Wn2 = (const float*)d_in[8];
  const float* bn2 = (const float*)d_in[9];
  const float* Ww  = (const float*)d_in[10];
  const float* bw  = (const float*)d_in[11];
  const float* Wm1 = (const float*)d_in[12];
  const float* bm1 = (const float*)d_in[13];
  const float* g1  = (const float*)d_in[14];
  const float* be1 = (const float*)d_in[15];
  const float* rm1 = (const float*)d_in[16];
  const float* rv1 = (const float*)d_in[17];
  const float* Wm2 = (const float*)d_in[18];
  const float* bm2 = (const float*)d_in[19];
  const float* g2  = (const float*)d_in[20];
  const float* be2 = (const float*)d_in[21];
  const float* rm2 = (const float*)d_in[22];
  const float* rv2 = (const float*)d_in[23];
  const float* Wm3 = (const float*)d_in[24];
  const float* bm3 = (const float*)d_in[25];

  const int N = in_sizes[0] / 64;
  const int E = in_sizes[1] / 3;
  const int nTiles = (N + 15) / 16;

  // Workspace layout (floats): accP[N*64] accN[N*64] cntP[N] cntN[N] z1[N*64]
  float* ws   = (float*)d_ws;
  float* accP = ws;
  float* accN = accP + (size_t)N * 64;
  float* cntP = accN + (size_t)N * 64;
  float* cntN = cntP + N;
  float* z1   = cntN + N;

  float* outZ    = (float*)d_out;
  float* outProb = outZ + (size_t)N * 64;

  // 1) zero accumulators + counts (deterministic each call)
  zero_kernel<<<grid_for((size_t)N * 130, 16384), 256, 0, stream>>>(
      accP, (size_t)N * 130);

  // 2) layer-1 signed mean-agg sums (x gathers, L2 float atomics)
  {
    const long long total = (long long)E * 16;
    edge_agg<<<(int)((total + 255) / 256), 256, 0, stream>>>(
        ei, x, accP, accN, cntP, cntN, E, 1);
  }

  // 3) layer-1 node GEMMs (WMMA bf16) -> z1 = [xp|xn]
  node_layer1<<<512, 128, 0, stream>>>(x, accP, accN, cntP, cntN, Wp1, bp1,
                                       Wn1, bn1, z1, N, nTiles);

  // 4) re-zero accumulators (counts are reused unchanged)
  zero_kernel<<<grid_for((size_t)N * 128, 16384), 256, 0, stream>>>(
      accP, (size_t)N * 128);

  // 5) layer-2 aggregation: identical scatter with z1 = [xp|xn] features
  {
    const long long total = (long long)E * 16;
    edge_agg<<<(int)((total + 255) / 256), 256, 0, stream>>>(
        ei, z1, accP, accN, cntP, cntN, E, 0);
  }

  // 6) fused layer-2 + projection + MLP head (WMMA) -> d_out
  node_layer2<<<1024, 64, 0, stream>>>(
      z1, accP, accN, cntP, cntN, Wp2, bp2, Wn2, bn2, Ww, bw, Wm1, bm1, g1,
      be1, rm1, rv1, Wm2, bm2, g2, be2, rm2, rv2, Wm3, bm3, outZ, outProb, N,
      nTiles);
}
